// MDmis_78735340470825
// MI455X (gfx1250) — matrix operand order repaired and checked by reading, not compiled, not run on hardware
//
#include <hip/hip_runtime.h>
#include <hip/hip_bf16.h>
#include <stdint.h>

typedef __attribute__((ext_vector_type(16))) __bf16 v16bf;
typedef __attribute__((ext_vector_type(8)))  float  v8f;
typedef __attribute__((ext_vector_type(2)))  __bf16 bf16x2;
typedef __attribute__((ext_vector_type(4)))  int    v4i;

#define B_   4
#define L_   512
#define HP_  128
#define MDP_ 10
#define DIN_ 256
#define H_   16
#define DQ_  64
#define D_   256
#define F_   1280

// ---------------------------------------------------------------------------
// CDNA5 async global->LDS copy (ASYNCcnt path), with safe fallback.
// Builtin signature (from compiler diagnostic): (int4* global_src,
// int4* lds_dst, imm offset, imm cpol).
// ---------------------------------------------------------------------------
#if defined(__AMDGCN__) && __has_builtin(__builtin_amdgcn_global_load_async_to_lds_b128) && \
    __has_builtin(__builtin_amdgcn_s_wait_asynccnt)
#define HAVE_ASYNC_LDS 1
typedef __attribute__((address_space(1))) v4i GInt4;
typedef __attribute__((address_space(3))) v4i LInt4;
#else
#define HAVE_ASYNC_LDS 0
#endif

__device__ __forceinline__ void cp16_async(void* lds_dst, const void* gsrc) {
#if HAVE_ASYNC_LDS
  __builtin_amdgcn_global_load_async_to_lds_b128(
      (GInt4*)(uintptr_t)gsrc,
      (LInt4*)(unsigned)(uintptr_t)lds_dst, 0, 0);
#else
  *(float4*)lds_dst = *(const float4*)gsrc;
#endif
}
__device__ __forceinline__ void cp_async_wait() {
#if HAVE_ASYNC_LDS
  __builtin_amdgcn_s_wait_asynccnt(0);
#endif
}

// ---------------------------------------------------------------------------
// Fold small weights: wsum[138], bsum, wf_fold[1280], cb[20], wd20_fold[256*20]
// ---------------------------------------------------------------------------
__global__ void fold_small_kernel(const float* __restrict__ W_pair,
                                  const float* __restrict__ b_pair,
                                  const float* __restrict__ W_f,
                                  const float* __restrict__ W_path,
                                  const float* __restrict__ b_d20,
                                  const float* __restrict__ b_f,
                                  const float* __restrict__ b_path,
                                  const float* __restrict__ W_d20,
                                  float* wsum, float* bsum, float* wf_fold,
                                  float* cb, float* wd20_fold) {
  int t = threadIdx.x;
  if (t < 138) {
    float s = 0.f;
    for (int p = 0; p < 32; ++p) s += W_pair[t * 32 + p];
    wsum[t] = s;
  }
  if (t == 138) {
    float s = 0.f;
    for (int p = 0; p < 32; ++p) s += b_pair[p];
    *bsum = s;
  }
  for (int f = t; f < F_; f += 256) {
    float s = 0.f;
    for (int d = 0; d < D_; ++d) s += W_f[f * D_ + d] * W_path[d];
    wf_fold[f] = s;
  }
  if (t < 20) {
    float s = 0.f;
    for (int d = 0; d < D_; ++d) s += (b_d20[d * 20 + t] + b_f[d]) * W_path[d];
    cb[t] = s + b_path[0];
  }
  for (int o = t; o < D_ * 20; o += 256) {
    int c = o / 20, k = o % 20;
    float s = 0.f;
    for (int d = 0; d < D_; ++d) s += W_d20[(size_t)c * (D_ * 20) + d * 20 + k] * W_path[d];
    wd20_fold[o] = s;
  }
}

// ---------------------------------------------------------------------------
// Fold Wo @ W_ld -> Wfold[1024x256] (bf16), bfold = bo @ W_ld + b_ld
// ---------------------------------------------------------------------------
__global__ void fold_wo_kernel(const float* __restrict__ Wo,
                               const float* __restrict__ bo,
                               const float* __restrict__ W_ld,
                               const float* __restrict__ b_ld,
                               __bf16* Wfold, float* bfold) {
  int n = blockIdx.x * blockDim.x + threadIdx.x;
  if (n < 1024 * 256) {
    int r = n / 256, c = n % 256;
    float s = 0.f;
    for (int j = 0; j < 512; ++j) s += Wo[(size_t)r * 512 + j] * W_ld[(size_t)j * 256 + c];
    Wfold[n] = (__bf16)s;
  }
  if (n < 256) {
    float s = b_ld[n];
    for (int j = 0; j < 512; ++j) s += bo[j] * W_ld[(size_t)j * 256 + n];
    bfold[n] = s;
  }
}

// ---------------------------------------------------------------------------
// Convert md_res -> bf16, pack [Wq|Wk|Wv] -> bf16 [256 x 3072]
// ---------------------------------------------------------------------------
__global__ void prep_bf_kernel(const float* __restrict__ md_res,
                               const float* __restrict__ Wq,
                               const float* __restrict__ Wk,
                               const float* __restrict__ Wv,
                               __bf16* Xbf, __bf16* Wqkv) {
  int i = blockIdx.x * blockDim.x + threadIdx.x;
  if (i < B_ * L_ * DIN_) Xbf[i] = (__bf16)md_res[i];
  if (i < DIN_ * 3072) {
    int k = i / 3072, n = i % 3072;
    float v;
    if (n < 1024)      v = Wq[(size_t)k * 1024 + n];
    else if (n < 2048) v = Wk[(size_t)k * 1024 + (n - 1024)];
    else               v = Wv[(size_t)k * 1024 + (n - 2048)];
    Wqkv[i] = (__bf16)v;
  }
}

// ---------------------------------------------------------------------------
// mask_bias[b,i,j] = pair . wsum + bsum + attention_masks  (578 MB stream)
// Wave per row, 4 rows per wave, prefetch ahead.
// ---------------------------------------------------------------------------
__global__ void bias_kernel(const float* __restrict__ pe,
                            const float* __restrict__ mdp,
                            const float* __restrict__ am,
                            const float* __restrict__ wsum,
                            const float* __restrict__ bsum,
                            float* __restrict__ mask_bias) {
  int wid = blockIdx.x * (blockDim.x >> 5) + (threadIdx.x >> 5);
  int lane = threadIdx.x & 31;
  const float4* w4 = (const float4*)wsum;
  float4 w = w4[lane];
  float wmd = wsum[HP_ + (lane & 15)];  // safe: wsum padded to 160
  float bs = *bsum;
#pragma unroll
  for (int it = 0; it < 4; ++it) {
    int gid = wid * 4 + it;
    const float* prow = pe + (size_t)gid * HP_;
    if (it < 3) __builtin_prefetch(prow + HP_ + lane * 4, 0, 1);  // next row
    float4 v = ((const float4*)prow)[lane];
    float s = v.x * w.x + v.y * w.y + v.z * w.z + v.w * w.w;
    if (lane < MDP_) s += mdp[(size_t)gid * MDP_ + lane] * wmd;
#pragma unroll
    for (int m = 16; m >= 1; m >>= 1) s += __shfl_xor(s, m, 32);
    if (lane == 0) mask_bias[gid] = s + bs + am[gid];
  }
}

// ---------------------------------------------------------------------------
// bf16 WMMA GEMM with async LDS staging:
// C[MxN](f32) = A[MxK](bf16,row) @ B[KxN](bf16,row) + bias
// Block: 128 thr / 4 waves; block tile 16(M) x 64(N); k-step 32.
// ---------------------------------------------------------------------------
__global__ __launch_bounds__(128)
void wmma_gemm_kernel(const __bf16* __restrict__ A, const __bf16* __restrict__ Bm,
                      const float* __restrict__ bias, float* __restrict__ C,
                      int M, int N, int K) {
  __shared__ __align__(16) __bf16 Abuf[16 * 32];   // 1 KB
  __shared__ __align__(16) __bf16 Bbuf[32 * 64];   // 4 KB
  int tid = threadIdx.x;
  int lane = tid & 31;
  int wave = tid >> 5;
  int m0 = blockIdx.x * 16;
  int nblk = blockIdx.y * 64;
  int n0 = nblk + wave * 16;

  int mA  = lane & 15;
  int hiA = (lane >> 4) << 4;   // +16 in K for lanes 16..31 (A layout)
  int nB  = lane & 15;
  int hiB = (lane >> 4) << 4;   // +16 in K for lanes 16..31 (B layout)

  v8f acc = {};
  for (int k0 = 0; k0 < K; k0 += 32) {
    // stage A(16x32)=64 chunks + B(32x64)=256 chunks of 16B
    for (int c = tid; c < 320; c += 128) {
      if (c < 64) {
        int row = c >> 2, seg = c & 3;
        cp16_async(Abuf + c * 8, A + (size_t)(m0 + row) * K + k0 + seg * 8);
      } else {
        int cc = c - 64;
        int row = cc >> 3, seg = cc & 7;
        cp16_async(Bbuf + cc * 8, Bm + (size_t)(k0 + row) * N + nblk + seg * 8);
      }
    }
    cp_async_wait();
    __syncthreads();

    v16bf a, b;
    const __bf16* Arow = Abuf + mA * 32 + hiA;
#pragma unroll
    for (int hp = 0; hp < 8; ++hp) {
      int kk2 = hp << 1;
      bf16x2 u = *(const bf16x2*)(Arow + kk2);
      a[2 * hp]     = u.x;
      a[2 * hp + 1] = u.y;
    }
#pragma unroll
    for (int h = 0; h < 16; ++h) {
      b[h] = Bbuf[(h + hiB) * 64 + wave * 16 + nB];
    }
    acc = __builtin_amdgcn_wmma_f32_16x16x32_bf16(false, a, false, b,
                                                  (short)0, acc, false, false);
    __syncthreads();
  }
  int mrow = (lane >> 4) << 3;
  float bv = bias ? bias[n0 + (lane & 15)] : 0.f;
#pragma unroll
  for (int r = 0; r < 8; ++r) {
    C[(size_t)(m0 + r + mrow) * N + n0 + (lane & 15)] = acc[r] + bv;
  }
}

// ---------------------------------------------------------------------------
// RoPE + layout [B,L,3072] f32 -> q/k/v [B,H,L,64] bf16 (rope on q,k)
// ---------------------------------------------------------------------------
__global__ void rope_kernel(const float* __restrict__ qkv,
                            __bf16* __restrict__ qb, __bf16* __restrict__ kb,
                            __bf16* __restrict__ vb) {
  int i = blockIdx.x * blockDim.x + threadIdx.x;
  if (i >= B_ * L_ * H_ * DQ_) return;
  int d = i & 63;
  int h = (i >> 6) & 15;
  int l = (i >> 10) & 511;
  int b = i >> 19;
  size_t src = ((size_t)(b * L_ + l)) * 3072 + h * 64 + d;
  size_t dst = ((size_t)((b * H_ + h) * L_) + l) * 64 + d;
  vb[dst] = (__bf16)qkv[src + 2048];
  if (d < 32) {
    float freq = __powf(10000.f, -(float)d * (1.f / 32.f));
    float ang = (float)l * freq;
    float sn, cs;
    __sincosf(ang, &sn, &cs);
    float q1 = qkv[src],        q2 = qkv[src + 32];
    float k1 = qkv[src + 1024], k2 = qkv[src + 1024 + 32];
    qb[dst]      = (__bf16)(q1 * cs - q2 * sn);
    qb[dst + 32] = (__bf16)(q1 * sn + q2 * cs);
    kb[dst]      = (__bf16)(k1 * cs - k2 * sn);
    kb[dst + 32] = (__bf16)(k1 * sn + k2 * cs);
  }
}

// ---------------------------------------------------------------------------
// Flash attention per (b,h). Block = 8 waves x 16 q-rows = 128 q rows.
// K/V tiles (32 keys) staged once per block via async-to-LDS; all waves share.
// Online softmax in wave32 half-groups; P transposed C->A layout via LDS.
// ---------------------------------------------------------------------------
__global__ __launch_bounds__(256)
void attn_kernel(const __bf16* __restrict__ Q, const __bf16* __restrict__ Kmat,
                 const __bf16* __restrict__ V, const float* __restrict__ mask_bias,
                 __bf16* __restrict__ O) {
  __shared__ __align__(16) __bf16 Ktile[32 * 64];   // 4 KB
  __shared__ __align__(16) __bf16 Vtile[32 * 64];   // 4 KB
  __shared__ float pshare[8][16 * 32];              // 16 KB
  int lane = threadIdx.x & 31;
  int wave = threadIdx.x >> 5;
  int bh = blockIdx.x;
  int b = bh >> 4;
  int hh = bh & 15;
  int q0 = (blockIdx.y * 8 + wave) * 16;

  const __bf16* Qb = Q + (size_t)bh * L_ * DQ_;
  const __bf16* Kb = Kmat + (size_t)bh * L_ * DQ_;
  const __bf16* Vb = V + (size_t)bh * L_ * DQ_;
  const float* mb = mask_bias + ((size_t)b * L_ + q0) * L_;

  int mA  = lane & 15;
  int hiA = (lane >> 4) << 4;
  int nB  = lane & 15;
  int hiB = (lane >> 4) << 4;
  int mrow = (lane >> 4) << 3;

  // Q fragments (16x64 = two 16x32 A tiles), loaded once
  v16bf qa0, qa1;
  const __bf16* Qrow = Qb + (size_t)(q0 + mA) * 64 + hiA;
#pragma unroll
  for (int hp = 0; hp < 8; ++hp) {
    int kk2 = hp << 1;
    bf16x2 u0 = *(const bf16x2*)(Qrow + kk2);
    bf16x2 u1 = *(const bf16x2*)(Qrow + kk2 + 32);
    qa0[2 * hp] = u0.x; qa0[2 * hp + 1] = u0.y;
    qa1[2 * hp] = u1.x; qa1[2 * hp + 1] = u1.y;
  }

  v8f oacc[4] = {{}, {}, {}, {}};
  float rmax[8], rsum[8];
#pragma unroll
  for (int r = 0; r < 8; ++r) { rmax[r] = -1e30f; rsum[r] = 0.f; }
  const float scale = 0.125f;  // 1/sqrt(64)
  float* pbuf = pshare[wave];

  for (int kb = 0; kb < L_; kb += 32) {
    // --- stage K,V rows kb..kb+31 into LDS (512 x 16B chunks, async) ---
    for (int c = threadIdx.x; c < 512; c += 256) {
      int which = c >> 8;
      int cc = c & 255;
      int row = cc >> 3, seg = cc & 7;
      const __bf16* src = (which ? Vb : Kb) + (size_t)(kb + row) * 64 + seg * 8;
      __bf16* dst = (which ? Vtile : Ktile) + cc * 8;
      cp16_async(dst, src);
    }
    cp_async_wait();
    __syncthreads();

    // --- S = Q @ K^T for 32 keys (two 16x16 tiles, 2 k-steps each) ---
    v16bf b0, b1;
    v8f s0 = {}, s1 = {};
#pragma unroll
    for (int h = 0; h < 16; ++h) {
      int dd = h + hiB;
      b0[h] = Ktile[nB * 64 + dd];
      b1[h] = Ktile[nB * 64 + dd + 32];
    }
    s0 = __builtin_amdgcn_wmma_f32_16x16x32_bf16(false, qa0, false, b0, (short)0, s0, false, false);
    s0 = __builtin_amdgcn_wmma_f32_16x16x32_bf16(false, qa1, false, b1, (short)0, s0, false, false);
#pragma unroll
    for (int h = 0; h < 16; ++h) {
      int dd = h + hiB;
      b0[h] = Ktile[(16 + nB) * 64 + dd];
      b1[h] = Ktile[(16 + nB) * 64 + dd + 32];
    }
    s1 = __builtin_amdgcn_wmma_f32_16x16x32_bf16(false, qa0, false, b0, (short)0, s1, false, false);
    s1 = __builtin_amdgcn_wmma_f32_16x16x32_bf16(false, qa1, false, b1, (short)0, s1, false, false);

    // --- online softmax per row ---
#pragma unroll
    for (int r = 0; r < 8; ++r) {
      int m = r + mrow;
      float v0 = s0[r] * scale + mb[(size_t)m * L_ + kb + (lane & 15)];
      float v1 = s1[r] * scale + mb[(size_t)m * L_ + kb + 16 + (lane & 15)];
      float t = fmaxf(v0, v1);
#pragma unroll
      for (int msk = 8; msk >= 1; msk >>= 1) t = fmaxf(t, __shfl_xor(t, msk, 32));
      float nm = fmaxf(rmax[r], t);
      float corr = __expf(rmax[r] - nm);
      rmax[r] = nm;
      float p0 = __expf(v0 - nm);
      float p1 = __expf(v1 - nm);
      float ps = p0 + p1;
#pragma unroll
      for (int msk = 8; msk >= 1; msk >>= 1) ps += __shfl_xor(ps, msk, 32);
      rsum[r] = rsum[r] * corr + ps;
      oacc[0][r] *= corr; oacc[1][r] *= corr; oacc[2][r] *= corr; oacc[3][r] *= corr;
      pbuf[m * 32 + (lane & 15)] = p0;
      pbuf[m * 32 + 16 + (lane & 15)] = p1;
    }
    __syncthreads();

    // --- read P back in A-matrix layout (16x32) ---
    v16bf pa;
    const float* Prow = pbuf + mA * 32 + hiA;
#pragma unroll
    for (int h = 0; h < 16; ++h) {
      int kk = h;
      pa[h] = (__bf16)Prow[kk];
    }

    // --- O += P @ V (4 N-tiles of 16 over DQ=64) ---
#pragma unroll
    for (int nt = 0; nt < 4; ++nt) {
      v16bf vfrag;
#pragma unroll
      for (int h = 0; h < 16; ++h) {
        vfrag[h] = Vtile[(h + hiB) * 64 + nt * 16 + nB];
      }
      oacc[nt] = __builtin_amdgcn_wmma_f32_16x16x32_bf16(false, pa, false, vfrag,
                                                         (short)0, oacc[nt], false, false);
    }
    __syncthreads();   // protect K/V/pshare tiles before next iteration
  }

  // --- normalize + write O as [B, L, H*DQ] bf16 ---
#pragma unroll
  for (int r = 0; r < 8; ++r) {
    float inv = 1.f / rsum[r];
    int m = r + mrow;
#pragma unroll
    for (int nt = 0; nt < 4; ++nt) {
      O[((size_t)(b * L_ + q0 + m)) * (H_ * DQ_) + hh * 64 + nt * 16 + (lane & 15)] =
          (__bf16)(oacc[nt][r] * inv);
    }
  }
}

// ---------------------------------------------------------------------------
// diffdot[bl,k] = sum_f diff_emb[bl,f,k] * wf_fold[f]   (210 MB stream)
// ---------------------------------------------------------------------------
__global__ void diffdot_kernel(const float* __restrict__ de,
                               const float* __restrict__ wf_fold,
                               float* __restrict__ diffdot) {
  __shared__ float acc[20];
  int bl = blockIdx.x;
  int t = threadIdx.x;
  if (t < 20) acc[t] = 0.f;
  __syncthreads();
  const float* base = de + (size_t)bl * F_ * 20;
  float loc[20];
#pragma unroll
  for (int k = 0; k < 20; ++k) loc[k] = 0.f;
  for (int f = t; f < F_; f += 256) {
    const float* row = base + (size_t)f * 20;
    __builtin_prefetch(row + 256 * 20, 0, 1);
    float w = wf_fold[f];
#pragma unroll
    for (int k = 0; k < 20; ++k) loc[k] += w * row[k];
  }
#pragma unroll
  for (int k = 0; k < 20; ++k) {
    float v = loc[k];
#pragma unroll
    for (int m = 16; m >= 1; m >>= 1) v += __shfl_xor(v, m, 32);
    if ((t & 31) == 0) atomicAdd(&acc[k], v);   // ds_add_f32
  }
  __syncthreads();
  if (t < 20) diffdot[bl * 20 + t] = acc[t];
}

// ---------------------------------------------------------------------------
// out[bl,k] = sigmoid( xd[bl,:] @ wd20_fold[:,k] + diffdot + cb )
// ---------------------------------------------------------------------------
__global__ void final_kernel(const float* __restrict__ xd,
                             const float* __restrict__ wd20_fold,
                             const float* __restrict__ diffdot,
                             const float* __restrict__ cb,
                             float* __restrict__ out) {
  __shared__ float xs[D_];
  int bl = blockIdx.x;
  int t = threadIdx.x;
  xs[t] = xd[(size_t)bl * D_ + t];
  __syncthreads();
  if (t < 20) {
    float s = diffdot[bl * 20 + t] + cb[t];
    for (int c = 0; c < D_; ++c) s += xs[c] * wd20_fold[c * 20 + t];
    out[bl * 20 + t] = 1.f / (1.f + __expf(-s));
  }
}

// ---------------------------------------------------------------------------
extern "C" void kernel_launch(void* const* d_in, const int* in_sizes, int n_in,
                              void* d_out, int out_size, void* d_ws, size_t ws_size,
                              hipStream_t stream) {
  const float* pair_emb = (const float*)d_in[0];
  const float* md_pair  = (const float*)d_in[1];
  const float* attn_m   = (const float*)d_in[2];
  const float* md_res   = (const float*)d_in[3];
  const float* diff_emb = (const float*)d_in[4];
  const float* W_pair   = (const float*)d_in[5];
  const float* b_pair   = (const float*)d_in[6];
  const float* Wq       = (const float*)d_in[7];
  const float* Wk       = (const float*)d_in[8];
  const float* Wv       = (const float*)d_in[9];
  const float* Wo       = (const float*)d_in[10];
  const float* bo       = (const float*)d_in[11];
  const float* W_ld     = (const float*)d_in[12];
  const float* b_ld     = (const float*)d_in[13];
  const float* W_d20    = (const float*)d_in[14];
  const float* b_d20    = (const float*)d_in[15];
  const float* W_f      = (const float*)d_in[16];
  const float* b_f      = (const float*)d_in[17];
  const float* W_path   = (const float*)d_in[18];
  const float* b_path   = (const float*)d_in[19];
  float* out = (float*)d_out;

  // workspace carve-up
  float* wsum      = (float*)d_ws;                 // 160 (pad)
  float* bsum      = wsum + 160;                   // 32 (pad)
  float* cb        = bsum + 32;                    // 32 (pad)
  float* wf_fold   = cb + 32;                      // 1280
  float* wd20_fold = wf_fold + 1280;               // 5120
  float* bfold     = wd20_fold + 5120;             // 256
  float* mask_bias = bfold + 256;                  // 1048576
  float* qkv       = mask_bias + (size_t)B_ * L_ * L_;       // 6291456
  float* xd        = qkv + (size_t)B_ * L_ * 3072;           // 524288
  float* diffdot   = xd + (size_t)B_ * L_ * D_;              // 40960
  __bf16* Xbf   = (__bf16*)(diffdot + 40960 + 64);
  __bf16* Wqkv  = Xbf + (size_t)B_ * L_ * DIN_;              // 786432
  __bf16* qb    = Wqkv + (size_t)DIN_ * 3072;
  __bf16* kbuf  = qb + (size_t)B_ * H_ * L_ * DQ_;
  __bf16* vbuf  = kbuf + (size_t)B_ * H_ * L_ * DQ_;
  __bf16* obf   = vbuf + (size_t)B_ * H_ * L_ * DQ_;
  __bf16* Wfold = obf + (size_t)B_ * L_ * H_ * DQ_;          // 262144

  fold_small_kernel<<<1, 256, 0, stream>>>(W_pair, b_pair, W_f, W_path, b_d20,
                                           b_f, b_path, W_d20,
                                           wsum, bsum, wf_fold, cb, wd20_fold);
  fold_wo_kernel<<<1024, 256, 0, stream>>>(Wo, bo, W_ld, b_ld, Wfold, bfold);
  prep_bf_kernel<<<3072, 256, 0, stream>>>(md_res, Wq, Wk, Wv, Xbf, Wqkv);

  // streaming bias: 1M rows, 4 rows per wave, 8 waves per block
  bias_kernel<<<(B_ * L_ * L_) / 32, 256, 0, stream>>>(pair_emb, md_pair, attn_m,
                                                       wsum, bsum, mask_bias);

  // QKV projection GEMM: [2048x256] @ [256x3072]
  wmma_gemm_kernel<<<dim3(2048 / 16, 3072 / 64), 128, 0, stream>>>(
      Xbf, Wqkv, nullptr, qkv, 2048, 3072, 256);

  rope_kernel<<<(B_ * L_ * H_ * DQ_) / 256, 256, 0, stream>>>(qkv, qb, kbuf, vbuf);

  // flash attention: grid (B*H, L/(16 q-rows * 8 waves))
  attn_kernel<<<dim3(B_ * H_, L_ / 128), 256, 0, stream>>>(qb, kbuf, vbuf,
                                                           mask_bias, obf);

  // folded (Wo @ W_ld) GEMM: [2048x1024] @ [1024x256] + bfold -> xd
  wmma_gemm_kernel<<<dim3(2048 / 16, 256 / 64), 128, 0, stream>>>(
      obf, Wfold, bfold, xd, 2048, 256, 1024);

  diffdot_kernel<<<B_ * L_, 256, 0, stream>>>(diff_emb, wf_fold, diffdot);
  final_kernel<<<B_ * L_, 256, 0, stream>>>(xd, wd20_fold, diffdot, cb, out);
}
